// DummyModel_89816356094558
// MI455X (gfx1250) — matrix operand order repaired and checked by reading, not compiled, not run on hardware
//
#include <hip/hip_runtime.h>

typedef _Float16 v16h __attribute__((ext_vector_type(16)));
typedef _Float16 v8h  __attribute__((ext_vector_type(8)));
typedef _Float16 v4h  __attribute__((ext_vector_type(4)));
typedef float    v8f  __attribute__((ext_vector_type(8)));
typedef int      v4i  __attribute__((ext_vector_type(4)));

#define WMMA_F16(A, B, C) \
  __builtin_amdgcn_wmma_f32_16x16x32_f16(false, (A), false, (B), (short)0, (C), false, false)

// Problem constants: B=2,H=16,S=2048,D=64
#define ATT_S   2048
#define ATT_D   64
#define ATT_BH  32
#define QTILE   64           // query rows per workgroup (4 waves x 16)
#define KBLK    64           // key/value rows per inner iteration
#define NT      (KBLK / 16)  // 16-col score tiles per iteration
#define NCH     (KBLK / 32)  // K=32 A-chunks of P per iteration
#define NKB     (ATT_S / KBLK)

#if defined(__has_builtin)
#if __has_builtin(__builtin_amdgcn_global_load_async_to_lds_b128)
#define USE_ASYNC 1
#endif
#endif
#ifndef USE_ASYNC
#define USE_ASYNC 0
#endif

#if __has_builtin(__builtin_amdgcn_s_wait_asynccnt)
#define WAIT_ASYNC(n) __builtin_amdgcn_s_wait_asynccnt(n)
#else
#define WAIT_ASYNC(n) asm volatile("s_wait_asynccnt %0" ::"i"(n) : "memory")
#endif

#if USE_ASYNC
// Each thread issues 8 b128 async copies per tensor (64x64 f32 block / 128 threads).
__device__ __forceinline__ void issue_block(const float* gK, const float* gV,
                                            float* sK, float* sV, int tid) {
#pragma unroll
  for (int i = 0; i < 8; ++i) {
    const int idx = (i * 128 + tid) * 4;
    __builtin_amdgcn_global_load_async_to_lds_b128((v4i*)(gK + idx), (v4i*)(sK + idx), 0, 0);
  }
#pragma unroll
  for (int i = 0; i < 8; ++i) {
    const int idx = (i * 128 + tid) * 4;
    __builtin_amdgcn_global_load_async_to_lds_b128((v4i*)(gV + idx), (v4i*)(sV + idx), 0, 0);
  }
}
#endif

__launch_bounds__(128)
__global__ void flash_attn_f16wmma(const float* __restrict__ Q,
                                   const float* __restrict__ K,
                                   const float* __restrict__ V,
                                   float* __restrict__ O)
{
  // f16 compute-layout arrays (padded strides, 16B-aligned vector access).
  __shared__ alignas(16) _Float16 Ksh[KBLK][ATT_D + 8];    // stride 144 B
  __shared__ alignas(16) _Float16 Vt[ATT_D][KBLK + 8];     // stride 144 B (transposed V)
  __shared__ alignas(16) _Float16 Pb[4][16][KBLK + 8];     // per-wave P tile
#if USE_ASYNC
  __shared__ alignas(16) float stageK[2][KBLK * ATT_D];    // raw f32 double buffers
  __shared__ alignas(16) float stageV[2][KBLK * ATT_D];
#endif

  const int tid  = threadIdx.x;
  const int w    = tid >> 5;
  const int lane = tid & 31;
  const int hi   = lane >> 4;     // 0: lanes 0-15, 1: lanes 16-31
  const int ln   = lane & 15;

  const int bh = blockIdx.x >> 5;
  const int qt = blockIdx.x & 31;
  const size_t base = (size_t)bh * ATT_S * ATT_D;
  const float* Qp = Q + base;
  const float* Kp = K + base;
  const float* Vp = V + base;
  float*       Op = O + base;

  // ---- Q rows -> WMMA A layout, f16, SCALE*log2(e) folded (exp2 domain) ----
  const float qscale = 0.2f * 1.4426950408889634f;
  const int qrow = qt * QTILE + w * 16 + ln;
  v16h aq0, aq1;
  {
    const float* qr = Qp + (size_t)qrow * ATT_D;
    const int d0 = hi * 8;                         // K = e + (e>=8?8:0) + 8*hi
#pragma unroll
    for (int e = 0; e < 8; ++e) aq0[e]     = (_Float16)(qr[d0 + e] * qscale);
#pragma unroll
    for (int e = 0; e < 8; ++e) aq0[e + 8] = (_Float16)(qr[d0 + 16 + e] * qscale);
#pragma unroll
    for (int e = 0; e < 8; ++e) aq1[e]     = (_Float16)(qr[32 + d0 + e] * qscale);
#pragma unroll
    for (int e = 0; e < 8; ++e) aq1[e + 8] = (_Float16)(qr[32 + d0 + 16 + e] * qscale);
  }

  // All-ones B matrix: rowsum(P) via WMMA instead of lane shuffles.
  v16h bones;
#pragma unroll
  for (int e = 0; e < 16; ++e) bones[e] = (_Float16)1.0f;

  float mrow[8], lrow[8];
  v8f acc[4];
#pragma unroll
  for (int r = 0; r < 8; ++r) { mrow[r] = -1e30f; lrow[r] = 0.0f; }
#pragma unroll
  for (int c = 0; c < 4; ++c) { v8f z = {}; acc[c] = z; }

#if USE_ASYNC
  issue_block(Kp, Vp, stageK[0], stageV[0], tid);
#endif

  for (int kb = 0; kb < NKB; ++kb) {
#if USE_ASYNC
    if (kb + 1 < NKB) {
      issue_block(Kp + (size_t)(kb + 1) * KBLK * ATT_D,
                  Vp + (size_t)(kb + 1) * KBLK * ATT_D,
                  stageK[(kb + 1) & 1], stageV[(kb + 1) & 1], tid);
      WAIT_ASYNC(16);     // block kb's 16 copies complete (in-order retire)
    } else {
      WAIT_ASYNC(0);
    }
#endif
    __syncthreads();      // prior compute done; staged data visible to all waves

    // ---- Convert staged f32 -> f16 compute layout (V transposed) ----
    {
#if USE_ASYNC
      const float* ksrc = &stageK[kb & 1][0];
      const float* vsrc = &stageV[kb & 1][0];
#else
      const float* ksrc = Kp + (size_t)kb * KBLK * ATT_D;
      const float* vsrc = Vp + (size_t)kb * KBLK * ATT_D;
#endif
#pragma unroll
      for (int i = 0; i < 8; ++i) {
        const int idx = (i * 128 + tid) * 4;
        const float4 k4 = *(const float4*)(ksrc + idx);
        const float4 v4 = *(const float4*)(vsrc + idx);
        const int row = idx >> 6;        // key row in block
        const int col = idx & 63;        // d (4 consecutive)
        v4h kh;
        kh[0] = (_Float16)k4.x; kh[1] = (_Float16)k4.y;
        kh[2] = (_Float16)k4.z; kh[3] = (_Float16)k4.w;
        *(v4h*)&Ksh[row][col] = kh;
        Vt[col + 0][row] = (_Float16)v4.x;
        Vt[col + 1][row] = (_Float16)v4.y;
        Vt[col + 2][row] = (_Float16)v4.z;
        Vt[col + 3][row] = (_Float16)v4.w;
      }
    }
    __syncthreads();

    // ---- Scores: NT 16x16 tiles, each Q(16x64) x K_tile^T(64x16) ----
    v8f sc[NT];
#pragma unroll
    for (int kt = 0; kt < NT; ++kt) {
      const _Float16* kr = &Ksh[kt * 16 + ln][hi * 8];    // B: K = e + (e>=8?8:0) + 8*hi
      v16h b0, b1;
      *(v8h*)&b0     = *(const v8h*)(kr);
      ((v8h*)&b0)[1] = *(const v8h*)(kr + 16);
      *(v8h*)&b1     = *(const v8h*)(kr + 32);
      ((v8h*)&b1)[1] = *(const v8h*)(kr + 48);
      v8f c = {};
      c = WMMA_F16(aq0, b0, c);
      c = WMMA_F16(aq1, b1, c);
      sc[kt] = c;
    }

    // ---- Online softmax: max via 4 shuffles/row; sum deferred to WMMA ----
#pragma unroll
    for (int r = 0; r < 8; ++r) {
      float mx = sc[0][r];
#pragma unroll
      for (int t = 1; t < NT; ++t) mx = fmaxf(mx, sc[t][r]);
#pragma unroll
      for (int off = 1; off < 16; off <<= 1)
        mx = fmaxf(mx, __shfl_xor(mx, off, 16));
      const float mnew = fmaxf(mrow[r], mx);
      const float sf = exp2f(mrow[r] - mnew);
      mrow[r] = mnew;
      lrow[r] *= sf;
#pragma unroll
      for (int c = 0; c < 4; ++c) acc[c][r] *= sf;
      const int M = r + 8 * hi;
#pragma unroll
      for (int t = 0; t < NT; ++t) {
        const float p = exp2f(sc[t][r] - mnew);
        Pb[w][M][t * 16 + ln] = (_Float16)p;   // C->A transpose via per-wave LDS
      }
    }

    // ---- Build P A-chunks; rowsum via ones-WMMA; P x V ----
    v16h ap[NCH];
#pragma unroll
    for (int ch = 0; ch < NCH; ++ch) {
      const _Float16* pr = &Pb[w][ln][ch * 32 + hi * 8];  // K = e + (e>=8?8:0) + 8*hi
      *(v8h*)&ap[ch]     = *(const v8h*)(pr);
      ((v8h*)&ap[ch])[1] = *(const v8h*)(pr + 16);
    }
    {
      v8f ls = {};
#pragma unroll
      for (int ch = 0; ch < NCH; ++ch) ls = WMMA_F16(ap[ch], bones, ls);
#pragma unroll
      for (int r = 0; r < 8; ++r) lrow[r] += ls[r];       // C layout matches lrow lanes
    }
#pragma unroll
    for (int c = 0; c < 4; ++c) {
#pragma unroll
      for (int ch = 0; ch < NCH; ++ch) {
        const _Float16* vr = &Vt[c * 16 + ln][ch * 32 + hi * 8]; // key=ch*32+e+(e>=8?8:0)+8*hi
        v16h bv;
        *(v8h*)&bv     = *(const v8h*)(vr);
        ((v8h*)&bv)[1] = *(const v8h*)(vr + 16);
        acc[c] = WMMA_F16(ap[ch], bv, acc[c]);
      }
    }
    // next iteration's top barrier protects Ksh/Vt before restaging
  }

  // ---- Epilogue: out = 2 * acc / l ----
  float rinv[8];
#pragma unroll
  for (int r = 0; r < 8; ++r) rinv[r] = 2.0f / lrow[r];
#pragma unroll
  for (int c = 0; c < 4; ++c) {
    const int col = c * 16 + ln;
#pragma unroll
    for (int r = 0; r < 8; ++r) {
      const int row = qt * QTILE + w * 16 + r + 8 * hi;
      Op[(size_t)row * ATT_D + col] = acc[c][r] * rinv[r];
    }
  }
}

extern "C" void kernel_launch(void* const* d_in, const int* in_sizes, int n_in,
                              void* d_out, int out_size, void* d_ws, size_t ws_size,
                              hipStream_t stream) {
  (void)in_sizes; (void)n_in; (void)out_size; (void)d_ws; (void)ws_size;
  const float* Q = (const float*)d_in[0];
  const float* K = (const float*)d_in[1];
  const float* V = (const float*)d_in[2];
  float* O = (float*)d_out;
  const int blocks = ATT_BH * (ATT_S / QTILE);   // 1024 workgroups
  flash_attn_f16wmma<<<blocks, 128, 0, stream>>>(Q, K, V, O);
}
